// SIGN_49572512530566
// MI455X (gfx1250) — matrix operand order repaired and checked by reading, not compiled, not run on hardware
//
#include <hip/hip_runtime.h>
#include <hip/hip_bf16.h>
#include <math.h>

// ---------------------------------------------------------------------------
// CDNA5 / gfx1250 fused GNN pipeline, round 3.
// - v_wmma_f32_16x16x32_f16 for all big GEMMs (f16 in, f32 accum)
// - 2x3 per-wave tile (B fragments reused across 2 M-tiles)
// - double-buffered, register-batched B staging (overlaps compute)
// - all fragments via ds_load_b128, padded LDS strides
// ---------------------------------------------------------------------------

typedef __attribute__((ext_vector_type(16))) _Float16 v16h;
typedef __attribute__((ext_vector_type(8)))  _Float16 v8h;
typedef __attribute__((ext_vector_type(8)))  float    v8f;

#define HID 192
#define CONV 32
#define RBFD 64
#define NT 36
#define BG 256
#define N_ATOMS 32768
#define E_A2A 262144
#define E_B2A 262144

#define SA_LD 200  // padded row stride (halfs) for 64xHID LDS tiles
#define RB_LD 72   // padded row stride for 64xRBFD rbf tile
#define BT_LD 40   // padded row stride for transposed B chunk [N][32]
#define BT_SZ (HID * BT_LD)

// ---- WMMA helpers (wave32; A and B both use contiguous k = 16*(lane/16) + s) ----------

__device__ __forceinline__ v8f wmma_f16(v16h a, v16h b, v8f c) {
  return __builtin_amdgcn_wmma_f32_16x16x32_f16(false, a, false, b,
                                                (short)0, c, false, false);
}

// A fragment: 16x32 f16 tile, row-major (lda % 8 == 0, base 16B-aligned).
__device__ __forceinline__ v16h load_a(const _Float16* A, int lda) {
  int lane = threadIdx.x & 31;
  int m  = lane & 15;
  int kh = (lane >> 4) * 16;
  const _Float16* base = A + m * lda + kh;
  v8h lo = *(const v8h*)(base);
  v8h hi = *(const v8h*)(base + 8);
  return __builtin_shufflevector(lo, hi, 0, 1, 2, 3, 4, 5, 6, 7,
                                 8, 9, 10, 11, 12, 13, 14, 15);
}

// B fragment from TRANSPOSED LDS chunk BsT[n][k], lane's 16 halfs contiguous.
__device__ __forceinline__ v16h load_bT(const _Float16* BsT, int ncol0) {
  int lane = threadIdx.x & 31;
  int n  = ncol0 + (lane & 15);
  int kb = (lane >> 4) * 16;
  const v8h* p = (const v8h*)(BsT + n * BT_LD + kb);
  v8h lo = p[0];
  v8h hi = p[1];
  return __builtin_shufflevector(lo, hi, 0, 1, 2, 3, 4, 5, 6, 7,
                                 8, 9, 10, 11, 12, 13, 14, 15);
}

// stage transposed-weight K-chunk (N=192): register-batched, 3 uint4/thread
__device__ __forceinline__ void stage_bT192(_Float16* BsT, const _Float16* WT,
                                            int Ktot, int kc) {
  int t = threadIdx.x;
  const _Float16* src = WT + kc * 32;
  uint4 r0, r1, r2;
  {
    int i = t;           // vector index: row n = i>>2, quarter j = i&3
    r0 = *(const uint4*)(src + (size_t)(i >> 2) * Ktot + (i & 3) * 8);
  }
  {
    int i = t + 256;
    r1 = *(const uint4*)(src + (size_t)(i >> 2) * Ktot + (i & 3) * 8);
  }
  {
    int i = t + 512;
    r2 = *(const uint4*)(src + (size_t)(i >> 2) * Ktot + (i & 3) * 8);
  }
  {
    int i = t;
    *(uint4*)(BsT + (i >> 2) * BT_LD + (i & 3) * 8) = r0;
  }
  {
    int i = t + 256;
    *(uint4*)(BsT + (i >> 2) * BT_LD + (i & 3) * 8) = r1;
  }
  {
    int i = t + 512;
    *(uint4*)(BsT + (i >> 2) * BT_LD + (i & 3) * 8) = r2;
  }
}

// stage transposed-weight K-chunk (N=32): threads < 128 copy one uint4
__device__ __forceinline__ void stage_bT32(_Float16* BsT, const _Float16* WT,
                                           int Ktot, int kc) {
  int t = threadIdx.x;
  if (t < 128) {
    int n = t >> 2, j = t & 3;
    *(uint4*)(BsT + n * BT_LD + j * 8) =
        *(const uint4*)(WT + (size_t)n * Ktot + kc * 32 + j * 8);
  }
}

// ---- small prep kernels ----------------------------------------------------

__global__ void k_f32_to_f16(const float* __restrict__ src,
                             _Float16* __restrict__ dst, int n) {
  int i = blockIdx.x * blockDim.x + threadIdx.x;
  if (i < n) dst[i] = (_Float16)src[i];
}

// transpose-convert: src[K][N] f32 -> dst[N][K] f16
__global__ void k_f32_to_f16_T(const float* __restrict__ src,
                               _Float16* __restrict__ dst, int K, int N) {
  int i = blockIdx.x * blockDim.x + threadIdx.x;
  if (i < K * N) {
    int k = i / N, n = i % N;
    dst[(size_t)n * K + k] = (_Float16)src[i];
  }
}

__global__ void k_zero(float* p, int n) {
  int i = blockIdx.x * blockDim.x + threadIdx.x;
  if (i < n) p[i] = 0.f;
}

// binfeat[bin][j] = relu(dist_emb[bin] . W_dist[:,j] + b_dist[j]); 4 x 192
__global__ void k_binfeat(const float* __restrict__ dist_emb,
                          const float* __restrict__ W_dist,
                          const float* __restrict__ b_dist,
                          float* __restrict__ binfeat) {
  int bin = blockIdx.x, j = threadIdx.x;
  float acc = b_dist[j];
  for (int k = 0; k < HID; ++k) acc += dist_emb[bin * HID + k] * W_dist[k * HID + j];
  binfeat[bin * HID + j] = fmaxf(acc, 0.f);
}

// ---- node GEMM: h = relu(node_emb[node_ids] @ W_node + b) -----------------
// 8 waves: mg = w&1 (2 M-tiles each), ng = w>>1 (3 N-tiles each, stride 4)

__global__ void __launch_bounds__(256)
k_node(const int* __restrict__ node_ids, const _Float16* __restrict__ emb16,
       const _Float16* __restrict__ WnT, const float* __restrict__ b_node,
       float* __restrict__ h32, _Float16* __restrict__ h16) {
  __shared__ __align__(16) _Float16 sA[64 * SA_LD];
  __shared__ __align__(16) _Float16 sBT[2][BT_SZ];
  int row0 = blockIdx.x * 64;
  int tid = threadIdx.x;

  // gather 64 embedding rows: row r = tid>>2, 6 vec4 chunks of 4-stride
  {
    int r = tid >> 2, c0 = tid & 3;
    int id = node_ids[row0 + r];
    const uint4* s = (const uint4*)(emb16 + (size_t)id * HID);
    uint4* d = (uint4*)(sA + r * SA_LD);
#pragma unroll
    for (int q = 0; q < 6; ++q) d[c0 + 4 * q] = s[c0 + 4 * q];
  }

  int w = tid >> 5, lane = tid & 31;
  int mg = w & 1, ng = w >> 1;
  v8f zero = {};
  v8f acc[2][3];
#pragma unroll
  for (int mi = 0; mi < 2; ++mi)
#pragma unroll
    for (int j = 0; j < 3; ++j) acc[mi][j] = zero;

  stage_bT192(sBT[0], WnT, HID, 0);
  __syncthreads();
  for (int kc = 0; kc < 6; ++kc) {  // K = 192
    const _Float16* cur = sBT[kc & 1];
    if (kc + 1 < 6) stage_bT192(sBT[(kc + 1) & 1], WnT, HID, kc + 1);
    v16h a0 = load_a(sA + (2 * mg) * 16 * SA_LD + kc * 32, SA_LD);
    v16h a1 = load_a(sA + (2 * mg + 1) * 16 * SA_LD + kc * 32, SA_LD);
#pragma unroll
    for (int j = 0; j < 3; ++j) {
      v16h b = load_bT(cur, (ng + 4 * j) * 16);
      acc[0][j] = wmma_f16(a0, b, acc[0][j]);
      acc[1][j] = wmma_f16(a1, b, acc[1][j]);
    }
    __syncthreads();
  }

  int nlo = lane & 15;
#pragma unroll
  for (int mi = 0; mi < 2; ++mi) {
    int m0 = (2 * mg + mi) * 16 + (lane >> 4) * 8;
#pragma unroll
    for (int j = 0; j < 3; ++j) {
      int col = (ng + 4 * j) * 16 + nlo;
      float bias = b_node[col];
#pragma unroll
      for (int i = 0; i < 8; ++i) {
        float v = fmaxf(acc[mi][j][i] + bias, 0.f);
        size_t idx = (size_t)(row0 + m0 + i) * HID + col;
        h32[idx] = v;
        h16[idx] = (_Float16)v;
      }
    }
  }
}

// ---- fused edge kernel -----------------------------------------------------

__global__ void __launch_bounds__(256)
k_edge(const float* __restrict__ edge_dist, const int* __restrict__ a2a_src,
       const int* __restrict__ a2a_dst, const float* __restrict__ mu,
       const float* __restrict__ binfeat, const _Float16* __restrict__ h16,
       const _Float16* __restrict__ WrbfT, const float* __restrict__ b_rbf,
       const _Float16* __restrict__ WaggT, const float* __restrict__ b_agg,
       const _Float16* __restrict__ Wfc1T, const float* __restrict__ b_fc1,
       float* __restrict__ hf) {
  __shared__ __align__(16) _Float16 sSrc[64 * SA_LD];
  __shared__ __align__(16) _Float16 sDst[64 * SA_LD];
  __shared__ __align__(16) _Float16 sEf[64 * SA_LD];
  __shared__ __align__(16) _Float16 sBond[64 * SA_LD];
  __shared__ __align__(16) _Float16 sRbf[64 * RB_LD];
  __shared__ __align__(16) _Float16 sBT[2][BT_SZ];
  __shared__ float sD[64];
  __shared__ float sMu[RBFD];
  __shared__ int sBin[64];

  int e0 = blockIdx.x * 64;
  int tid = threadIdx.x;

  if (tid < 64) {
    float d = edge_dist[e0 + tid];
    sD[tid] = d;
    float dc = fminf(fmaxf(d, 1.0f), 4.99999f);
    sBin[tid] = (int)dc - 1;
    sMu[tid] = mu[tid];
  }
  // gather h16 rows for src/dst: row r = tid>>2, 6 vec4 chunks each
  {
    int r = tid >> 2, c0 = tid & 3;
    int si = a2a_src[e0 + r];
    int di = a2a_dst[e0 + r];
    const uint4* ps = (const uint4*)(h16 + (size_t)si * HID);
    const uint4* pd = (const uint4*)(h16 + (size_t)di * HID);
    uint4* ds_ = (uint4*)(sSrc + r * SA_LD);
    uint4* dd_ = (uint4*)(sDst + r * SA_LD);
#pragma unroll
    for (int q = 0; q < 6; ++q) {
      int c = c0 + 4 * q;
      ds_[c] = ps[c];
      dd_[c] = pd[c];
    }
  }
  __syncthreads();

  // rbf[e][k] = exp(-64*(exp(-d_e)-mu_k)^2)
  for (int i = tid; i < 64 * RBFD; i += 256) {
    int e = i >> 6, k = i & 63;
    float ed = __expf(-sD[e]);
    float t = ed - sMu[k];
    sRbf[e * RB_LD + k] = (_Float16)__expf(-64.0f * t * t);
  }

  int w = tid >> 5, lane = tid & 31;
  int mg = w & 1, ng = w >> 1;
  int nlo = lane & 15;
  v8f zero = {};
  v8f acc[2][3];

  // ---- GEMM1: ef = binfeat[bin] + relu(rbf @ W_rbf + b_rbf), K=64 ----
#pragma unroll
  for (int mi = 0; mi < 2; ++mi)
#pragma unroll
    for (int j = 0; j < 3; ++j) acc[mi][j] = zero;
  stage_bT192(sBT[0], WrbfT, RBFD, 0);
  __syncthreads();
  for (int kc = 0; kc < 2; ++kc) {
    const _Float16* cur = sBT[kc & 1];
    if (kc + 1 < 2) stage_bT192(sBT[(kc + 1) & 1], WrbfT, RBFD, kc + 1);
    v16h a0 = load_a(sRbf + (2 * mg) * 16 * RB_LD + kc * 32, RB_LD);
    v16h a1 = load_a(sRbf + (2 * mg + 1) * 16 * RB_LD + kc * 32, RB_LD);
#pragma unroll
    for (int j = 0; j < 3; ++j) {
      v16h b = load_bT(cur, (ng + 4 * j) * 16);
      acc[0][j] = wmma_f16(a0, b, acc[0][j]);
      acc[1][j] = wmma_f16(a1, b, acc[1][j]);
    }
    __syncthreads();
  }
#pragma unroll
  for (int mi = 0; mi < 2; ++mi) {
    int m0 = (2 * mg + mi) * 16 + (lane >> 4) * 8;
#pragma unroll
    for (int j = 0; j < 3; ++j) {
      int col = (ng + 4 * j) * 16 + nlo;
      float bias = b_rbf[col];
#pragma unroll
      for (int i = 0; i < 8; ++i) {
        int m = m0 + i;
        float v = fmaxf(acc[mi][j][i] + bias, 0.f) + binfeat[sBin[m] * HID + col];
        sEf[m * SA_LD + col] = (_Float16)v;
      }
    }
  }

  // ---- GEMM2: bond = relu(concat[src,dst,ef] @ W_agg + b_agg), K=576 ----
#pragma unroll
  for (int mi = 0; mi < 2; ++mi)
#pragma unroll
    for (int j = 0; j < 3; ++j) acc[mi][j] = zero;
  stage_bT192(sBT[0], WaggT, 3 * HID, 0);
  __syncthreads();
  for (int kc = 0; kc < 18; ++kc) {
    const _Float16* cur = sBT[kc & 1];
    if (kc + 1 < 18) stage_bT192(sBT[(kc + 1) & 1], WaggT, 3 * HID, kc + 1);
    const _Float16* Asrc;
    int ko;
    if (kc < 6)       { Asrc = sSrc; ko = kc; }
    else if (kc < 12) { Asrc = sDst; ko = kc - 6; }
    else              { Asrc = sEf;  ko = kc - 12; }
    v16h a0 = load_a(Asrc + (2 * mg) * 16 * SA_LD + ko * 32, SA_LD);
    v16h a1 = load_a(Asrc + (2 * mg + 1) * 16 * SA_LD + ko * 32, SA_LD);
#pragma unroll
    for (int j = 0; j < 3; ++j) {
      v16h b = load_bT(cur, (ng + 4 * j) * 16);
      acc[0][j] = wmma_f16(a0, b, acc[0][j]);
      acc[1][j] = wmma_f16(a1, b, acc[1][j]);
    }
    __syncthreads();
  }
#pragma unroll
  for (int mi = 0; mi < 2; ++mi) {
    int m0 = (2 * mg + mi) * 16 + (lane >> 4) * 8;
#pragma unroll
    for (int j = 0; j < 3; ++j) {
      int col = (ng + 4 * j) * 16 + nlo;
      float bias = b_agg[col];
#pragma unroll
      for (int i = 0; i < 8; ++i) {
        float v = fmaxf(acc[mi][j][i] + bias, 0.f);
        sBond[(m0 + i) * SA_LD + col] = (_Float16)v;
      }
    }
  }

  // ---- GEMM3: hf = relu(bond @ W_fc1 + b_fc1), K=192, N=32 ----
  // one 16x16 tile per wave: rows mt3 = w&3, cols nt3 = w>>2
  int mt3 = w & 3, nt3 = w >> 2;
  v8f c3 = zero;
  stage_bT32(sBT[0], Wfc1T, HID, 0);
  __syncthreads();
  for (int kc = 0; kc < 6; ++kc) {
    const _Float16* cur = sBT[kc & 1];
    if (kc + 1 < 6) stage_bT32(sBT[(kc + 1) & 1], Wfc1T, HID, kc + 1);
    v16h a = load_a(sBond + mt3 * 16 * SA_LD + kc * 32, SA_LD);
    v16h b = load_bT(cur, nt3 * 16);
    c3 = wmma_f16(a, b, c3);
    __syncthreads();
  }
  {
    int col = nt3 * 16 + nlo;
    int m0 = mt3 * 16 + (lane >> 4) * 8;
    float bias = b_fc1[col];
#pragma unroll
    for (int i = 0; i < 8; ++i) {
      float v = fmaxf(c3[i] + bias, 0.f);
      hf[(size_t)(e0 + m0 + i) * CONV + col] = v;
    }
  }
}

// ---- segment sums (seg ids sorted -> low atomic contention) ---------------

__global__ void k_scatter_b2a(const int* __restrict__ b2a_src,
                              const int* __restrict__ b2a_seg,
                              const float* __restrict__ hf,
                              float* __restrict__ gsum) {
  int i = blockIdx.x * blockDim.x + threadIdx.x;
  if (i >= E_B2A * 32) return;
  int e = i >> 5, c = i & 31;
  atomicAdd(&gsum[(size_t)b2a_seg[e] * CONV + c],
            hf[(size_t)b2a_src[e] * CONV + c]);
}

__global__ void k_atom_seg(const int* __restrict__ atom_seg,
                           const float* __restrict__ h32,
                           float* __restrict__ gfeat) {
  int a = blockIdx.x, c = threadIdx.x;  // 32768 blocks x 192 threads
  atomicAdd(&gfeat[(size_t)atom_seg[a] * HID + c], h32[(size_t)a * HID + c]);
}

// ---- bond scores: gv = gsum @ w_bo, mask, softmax over 36 -----------------

__global__ void k_bond_scores(const float* __restrict__ gsum,
                              const float* __restrict__ w_bo,
                              const int* __restrict__ mask,
                              float* __restrict__ out_bond) {
  int b = blockIdx.x, t = threadIdx.x;  // 256 blocks x 64 threads
  __shared__ float gv[NT];
  __shared__ float smx, ssum;
  if (t < NT) {
    const float* gs = gsum + (size_t)(b * NT + t) * CONV;
    float acc = 0.f;
    for (int k = 0; k < CONV; ++k) acc += gs[k] * w_bo[k];
    gv[t] = mask[b * NT + t] ? -1e9f : acc;
  }
  __syncthreads();
  if (t == 0) {
    float M = -INFINITY;
    for (int i = 0; i < NT; ++i) M = fmaxf(M, gv[i]);
    float S = 0.f;
    for (int i = 0; i < NT; ++i) S += __expf(gv[i] - M);
    smx = M;
    ssum = S;
  }
  __syncthreads();
  if (t < NT) out_bond[b * NT + t] = __expf(gv[t] - smx) / ssum;
}

// ---- dense output head: per-graph 192->256->256 MLP + two heads -----------

__global__ void __launch_bounds__(256)
k_mlp(const float* __restrict__ gfeat, const float* __restrict__ W_m1,
      const float* __restrict__ b_m1, const float* __restrict__ W_m2,
      const float* __restrict__ b_m2, const float* __restrict__ w_pred,
      const float* __restrict__ b_pred, const float* __restrict__ W_sc,
      const float* __restrict__ b_sc, float* __restrict__ out) {
  int b = blockIdx.x, t = threadIdx.x;  // 256 blocks x 256 threads
  __shared__ float x[HID], y1[256], y2[256];
  if (t < HID) x[t] = gfeat[b * HID + t];
  __syncthreads();
  float a = b_m1[t];
  for (int k = 0; k < HID; ++k) a += x[k] * W_m1[k * 256 + t];
  y1[t] = fmaxf(a, 0.f);
  __syncthreads();
  float a2 = b_m2[t];
  for (int k = 0; k < 256; ++k) a2 += y1[k] * W_m2[k * 256 + t];
  y2[t] = fmaxf(a2, 0.f);
  __syncthreads();
  if (t == 0) {
    float p = b_pred[0];
    for (int k = 0; k < 256; ++k) p += y2[k] * w_pred[k];
    out[b] = p;  // pred
  }
  if (t < NT) {
    float s = b_sc[t];
    for (int k = 0; k < 256; ++k) s += y2[k] * W_sc[k * NT + t];
    out[BG + b * NT + t] = s;  // atom_scores
  }
}

// ---------------------------------------------------------------------------

extern "C" void kernel_launch(void* const* d_in, const int* in_sizes, int n_in,
                              void* d_out, int out_size, void* d_ws,
                              size_t ws_size, hipStream_t stream) {
  const int*   node_ids  = (const int*)d_in[0];
  const float* edge_dist = (const float*)d_in[1];
  const int*   a2a_src   = (const int*)d_in[2];
  const int*   a2a_dst   = (const int*)d_in[3];
  const int*   atom_seg  = (const int*)d_in[4];
  const int*   b2a_src   = (const int*)d_in[5];
  const int*   b2a_seg   = (const int*)d_in[6];
  const int*   mask_mat  = (const int*)d_in[7];
  const float* node_emb  = (const float*)d_in[8];
  const float* W_node    = (const float*)d_in[9];
  const float* b_node    = (const float*)d_in[10];
  const float* dist_emb  = (const float*)d_in[11];
  const float* W_dist    = (const float*)d_in[12];
  const float* b_dist    = (const float*)d_in[13];
  const float* mu        = (const float*)d_in[14];
  const float* W_rbf     = (const float*)d_in[15];
  const float* b_rbf     = (const float*)d_in[16];
  const float* W_agg     = (const float*)d_in[17];
  const float* b_agg     = (const float*)d_in[18];
  const float* W_fc1     = (const float*)d_in[19];
  const float* b_fc1     = (const float*)d_in[20];
  const float* w_bo      = (const float*)d_in[21];
  const float* W_m1      = (const float*)d_in[22];
  const float* b_m1      = (const float*)d_in[23];
  const float* W_m2      = (const float*)d_in[24];
  const float* b_m2      = (const float*)d_in[25];
  const float* w_pred    = (const float*)d_in[26];
  const float* b_pred    = (const float*)d_in[27];
  const float* W_sc      = (const float*)d_in[28];
  const float* b_sc      = (const float*)d_in[29];

  float* out = (float*)d_out;  // [pred 256 | atom 9216 | bond 9216]

  size_t off = 0;
  auto take = [&](size_t bytes) -> char* {
    off = (off + 255) & ~(size_t)255;
    char* p = (char*)d_ws + off;
    off += bytes;
    return p;
  };
  float*    h32     = (float*)take((size_t)N_ATOMS * HID * 4);
  _Float16* h16     = (_Float16*)take((size_t)N_ATOMS * HID * 2);
  float*    hf      = (float*)take((size_t)E_A2A * CONV * 4);
  _Float16* emb16   = (_Float16*)take((size_t)64 * HID * 2);
  _Float16* WnT     = (_Float16*)take((size_t)HID * HID * 2);
  _Float16* WrbfT   = (_Float16*)take((size_t)RBFD * HID * 2);
  _Float16* WaggT   = (_Float16*)take((size_t)3 * HID * HID * 2);
  _Float16* Wfc1T   = (_Float16*)take((size_t)HID * CONV * 2);
  float*    binfeat = (float*)take((size_t)4 * HID * 4);
  float*    gsum    = (float*)take((size_t)BG * NT * CONV * 4);
  float*    gfeat   = (float*)take((size_t)BG * HID * 4);
  (void)ws_size;

  // 1) weight conversion (B matrices transposed) + per-bin dist features
  k_f32_to_f16<<<(64 * HID + 255) / 256, 256, 0, stream>>>(node_emb, emb16,
                                                           64 * HID);
  auto cvtT = [&](const float* s, _Float16* d, int K, int N) {
    k_f32_to_f16_T<<<(K * N + 255) / 256, 256, 0, stream>>>(s, d, K, N);
  };
  cvtT(W_node, WnT, HID, HID);
  cvtT(W_rbf, WrbfT, RBFD, HID);
  cvtT(W_agg, WaggT, 3 * HID, HID);
  cvtT(W_fc1, Wfc1T, HID, CONV);
  k_binfeat<<<4, HID, 0, stream>>>(dist_emb, W_dist, b_dist, binfeat);

  // 2) zero accumulators
  k_zero<<<(BG * NT * CONV + 255) / 256, 256, 0, stream>>>(gsum, BG * NT * CONV);
  k_zero<<<(BG * HID + 255) / 256, 256, 0, stream>>>(gfeat, BG * HID);

  // 3) node features (WMMA GEMM)
  k_node<<<N_ATOMS / 64, 256, 0, stream>>>(node_ids, emb16, WnT, b_node, h32, h16);

  // 4) fused edge pipeline (3 WMMA GEMMs, K=576 agg stage)
  k_edge<<<E_A2A / 64, 256, 0, stream>>>(edge_dist, a2a_src, a2a_dst, mu,
                                         binfeat, h16, WrbfT, b_rbf, WaggT,
                                         b_agg, Wfc1T, b_fc1, hf);

  // 5) segment reductions
  k_scatter_b2a<<<(E_B2A * 32) / 256, 256, 0, stream>>>(b2a_src, b2a_seg, hf, gsum);
  k_atom_seg<<<N_ATOMS, HID, 0, stream>>>(atom_seg, h32, gfeat);

  // 6) heads
  k_bond_scores<<<BG, 64, 0, stream>>>(gsum, w_bo, mask_mat, out + BG + BG * NT);
  k_mlp<<<BG, 256, 0, stream>>>(gfeat, W_m1, b_m1, W_m2, b_m2, w_pred, b_pred,
                                W_sc, b_sc, out);
}
